// SS2D_824633721058
// MI455X (gfx1250) — compile-verified, matrix operand order checked by reading, not run on hardware
//
#include <hip/hip_runtime.h>
#include <hip/hip_bf16.h>

typedef __attribute__((ext_vector_type(16))) _Float16 v16h;
typedef __attribute__((ext_vector_type(8)))  float    v8f;

// Problem constants (fixed by the reference's setup_inputs)
#define DCH   192          // d_inner
#define RNK   6            // dt_rank
#define NST   16           // d_state
#define MROW  224          // 192 (W_delta) + 32 (B,C projection rows)
#define LSEQ  4096         // max_bin
#define BSZ   4            // number of clouds
#define BM    (BSZ*LSEQ)   // 16384 padded positions
#define KT    (DCH/32)     // 6 k-steps of 32
#define NTPW  4            // N-tiles per wave in the GEMM
#define WVS   4            // waves per GEMM block

// Fragment-slot mapping for 16-bit WMMA operands (ISA 7.12.2):
// element (row/col, kk in 0..31) lives at lane = rc + 16*((kk&15)>=8),
// slot j = 8*(kk>=16) + (kk&7).  Fragment storage: [tile][kt][lane][16].
__device__ __forceinline__ void frag_slot(int rc, int kk, int& lane, int& j) {
    lane = rc + (((kk & 15) >= 8) ? 16 : 0);
    j    = ((kk >> 4) << 3) | (kk & 7);
}

// ---------------------------------------------------------------------------
// K0: build Wall (224 x 192) in A-fragment layout:
//   rows 0..191   : W_delta[d][j] = sum_r dt_w[d][r] * xpw[r][j]
//   rows 192..223 : xpw[6..37][j]   (B then C projection rows)
// ---------------------------------------------------------------------------
__global__ void k_prep_wall(const float* __restrict__ xpw,
                            const float* __restrict__ dtw,
                            _Float16* __restrict__ wall_sw) {
    int idx = blockIdx.x * blockDim.x + threadIdx.x;
    if (idx >= MROW * DCH) return;
    int m = idx / DCH, k = idx - m * DCH;
    float w;
    if (m < DCH) {
        w = 0.f;
        #pragma unroll
        for (int r = 0; r < RNK; ++r)
            w += dtw[m * RNK + r] * xpw[r * DCH + k];
    } else {
        w = xpw[(m - DCH + RNK) * DCH + k];
    }
    int mtile = m >> 4, row = m & 15, kt = k >> 5, kk = k & 31;
    int lane, j; frag_slot(row, kk, lane, j);
    wall_sw[((size_t)mtile * KT + kt) * 512 + lane * 16 + j] = (_Float16)w;
}

// ---------------------------------------------------------------------------
// K1: gather xs[d][p] = x[d][ order[padded_idx[p]] ]
//   -> linear f16 (for conv) + B-fragment-swizzled f16 (for WMMA)
// grid: (BM/256, 192)
// ---------------------------------------------------------------------------
__global__ void k_gather(const float* __restrict__ x,
                         const int* __restrict__ order,
                         const int* __restrict__ padded_idx,
                         _Float16* __restrict__ xs_lin,
                         _Float16* __restrict__ xs_sw,
                         int total) {
    int p = blockIdx.x * blockDim.x + threadIdx.x;
    int d = blockIdx.y;
    int g = order[padded_idx[p]];
    _Float16 v = (_Float16)x[(size_t)d * total + g];
    xs_lin[(size_t)d * BM + p] = v;
    int t = p >> 4, col = p & 15, kt = d >> 5, kk = d & 31;
    int lane, j; frag_slot(col, kk, lane, j);
    xs_sw[((size_t)t * KT + kt) * 512 + lane * 16 + j] = v;
}

// ---------------------------------------------------------------------------
// K2: WMMA GEMM  dbl(224 x BM) = Wall(224 x 192) * xs(192 x BM)
//   rows 0..191 = delta_raw ; rows 192..207 = B ; rows 208..223 = C
// Each wave: 1 M-tile x 4 N-tiles.  ALL operand fragments (6 A + 24 B)
// loaded up front; __launch_bounds__(128, 1) lifts the scheduler's
// occupancy-driven register-pressure cap so the 48 b128 loads can stay
// in flight instead of being re-sunk to load->wait(0)->wmma pairs.
// grid: (BM/(16*NTPW*WVS)=64, 14), block 128.
// ---------------------------------------------------------------------------
__global__ void __launch_bounds__(WVS * 32, 1)
k_wmma_gemm(const _Float16* __restrict__ wall_sw,
            const _Float16* __restrict__ xs_sw,
            float* __restrict__ dbl) {
    const int lane   = threadIdx.x & 31;
    const int wave   = threadIdx.x >> 5;
    const int mtile  = blockIdx.y;
    const int ntbase = (blockIdx.x * WVS + wave) * NTPW;

    // Issue every operand load up front (distinct, simultaneously-live regs)
    v16h a[KT];
    const _Float16* Ap = wall_sw + (size_t)mtile * KT * 512 + lane * 16;
    #pragma unroll
    for (int kt = 0; kt < KT; ++kt)
        a[kt] = *(const v16h*)(Ap + kt * 512);

    v16h b[KT * NTPW];
    #pragma unroll
    for (int nt = 0; nt < NTPW; ++nt) {
        const _Float16* Bp = xs_sw + (size_t)(ntbase + nt) * KT * 512 + lane * 16;
        #pragma unroll
        for (int kt = 0; kt < KT; ++kt)
            b[kt * NTPW + nt] = *(const v16h*)(Bp + kt * 512);
    }

    v8f acc[NTPW];
    #pragma unroll
    for (int nt = 0; nt < NTPW; ++nt) { v8f z = {}; acc[nt] = z; }

    #pragma unroll
    for (int kt = 0; kt < KT; ++kt)
        #pragma unroll
        for (int nt = 0; nt < NTPW; ++nt)
            acc[nt] = __builtin_amdgcn_wmma_f32_16x16x32_f16(
                          false, a[kt], false, b[kt * NTPW + nt], (short)0,
                          acc[nt], false, false);

    const int colN = lane & 15;
    const int moff = (lane >> 4) * 8;
    #pragma unroll
    for (int nt = 0; nt < NTPW; ++nt) {
        int n0 = (ntbase + nt) * 16;
        #pragma unroll
        for (int v = 0; v < 8; ++v)
            dbl[(size_t)(mtile * 16 + moff + v) * BM + n0 + colN] = acc[nt][v];
    }
}

// ---------------------------------------------------------------------------
// K3: depthwise conv3 (+bias) on xs -> u, and in-place softplus(delta + bias)
// grid: (BM/256, 192)
// ---------------------------------------------------------------------------
__global__ void k_conv_softplus(const _Float16* __restrict__ xs_lin,
                                const float* __restrict__ conv_w,
                                const float* __restrict__ conv_b,
                                const float* __restrict__ dt_bias,
                                float* __restrict__ dbl,   // rows 0..191 in-place
                                float* __restrict__ u) {
    int p = blockIdx.x * blockDim.x + threadIdx.x;
    int d = blockIdx.y;
    int l = p & (LSEQ - 1);
    size_t o = (size_t)d * BM + p;

    float xm1 = (l > 0)        ? (float)xs_lin[o - 1] : 0.f;
    float x0  = (float)xs_lin[o];
    float xp1 = (l < LSEQ - 1) ? (float)xs_lin[o + 1] : 0.f;
    u[o] = conv_w[d * 3 + 0] * xm1 + conv_w[d * 3 + 1] * x0 +
           conv_w[d * 3 + 2] * xp1 + conv_b[d];

    float t = dbl[o] + dt_bias[d];
    dbl[o] = (t > 20.f) ? t : log1pf(__expf(t));   // softplus, overflow-guarded
}

// ---------------------------------------------------------------------------
// K4: selective scan. One sequence (b,d) per 16-lane group; lane = state n.
//   h = exp(delta*A[d][n])*h + delta*B[b][n][l]*u ;  y = sum_n h*C + D*u
// ---------------------------------------------------------------------------
__global__ void k_scan(const float* __restrict__ dbl,
                       const float* __restrict__ u,
                       const float* __restrict__ A_logs,
                       const float* __restrict__ Ds,
                       float* __restrict__ ys) {
    int gtid = blockIdx.x * blockDim.x + threadIdx.x;
    int seq  = gtid >> 4;
    int n    = gtid & 15;
    if (seq >= BSZ * DCH) return;
    int b = seq / DCH;
    int d = seq - b * DCH;

    const float A  = -__expf(A_logs[d * NST + n]);
    const float Dd = Ds[d];
    const size_t base = (size_t)b * LSEQ;
    const float* drow = dbl + (size_t)d * BM + base;
    const float* urow = u   + (size_t)d * BM + base;
    const float* Brow = dbl + (size_t)(DCH + n) * BM + base;
    const float* Crow = dbl + (size_t)(DCH + NST + n) * BM + base;
    float* yrow = ys + (size_t)d * BM + base;

    float h = 0.f;
    for (int l = 0; l < LSEQ; ++l) {
        if ((l & 31) == 0) {   // stream-ahead prefetch (global_prefetch_b8)
            __builtin_prefetch(drow + l + 128, 0, 1);
            __builtin_prefetch(urow + l + 128, 0, 1);
            __builtin_prefetch(Brow + l + 128, 0, 1);
            __builtin_prefetch(Crow + l + 128, 0, 1);
        }
        float dt = drow[l];
        float uu = urow[l];
        h = __expf(dt * A) * h + dt * Brow[l] * uu;
        float yc = h * Crow[l];
        #pragma unroll
        for (int off = 8; off >= 1; off >>= 1)
            yc += __shfl_xor(yc, off, 32);   // reduce within 16-lane group
        if (n == 0) yrow[l] = yc + Dd * uu;
    }
}

// ---------------------------------------------------------------------------
// K5: crop valid, inverse scatter, transpose, LayerNorm(192). One wave/point.
// ---------------------------------------------------------------------------
__global__ void k_finalize(const float* __restrict__ ys,
                           const int* __restrict__ inverse,
                           const int* __restrict__ valid_idx,
                           const float* __restrict__ gamma,
                           const float* __restrict__ beta,
                           float* __restrict__ out, int total) {
    int tid  = blockIdx.x * blockDim.x + threadIdx.x;
    int pt   = tid >> 5;
    int lane = tid & 31;
    if (pt >= total) return;
    int p = valid_idx[inverse[pt]];

    float v[6], s = 0.f, s2 = 0.f;
    #pragma unroll
    for (int q = 0; q < 6; ++q) {
        int d = lane + 32 * q;
        v[q] = ys[(size_t)d * BM + p];
        s  += v[q];
        s2 += v[q] * v[q];
    }
    #pragma unroll
    for (int off = 16; off >= 1; off >>= 1) {
        s  += __shfl_xor(s,  off, 32);
        s2 += __shfl_xor(s2, off, 32);
    }
    float mu  = s * (1.f / DCH);
    float var = s2 * (1.f / DCH) - mu * mu;
    float inv = rsqrtf(var + 1e-5f);
    #pragma unroll
    for (int q = 0; q < 6; ++q) {
        int d = lane + 32 * q;
        out[(size_t)pt * DCH + d] = (v[q] - mu) * inv * gamma[d] + beta[d];
    }
}

// ---------------------------------------------------------------------------
extern "C" void kernel_launch(void* const* d_in, const int* in_sizes, int n_in,
                              void* d_out, int out_size, void* d_ws, size_t ws_size,
                              hipStream_t stream) {
    const float* x        = (const float*)d_in[0];
    const float* xpw      = (const float*)d_in[1];
    const float* dtw      = (const float*)d_in[2];
    const float* dt_bias  = (const float*)d_in[3];
    const float* A_logs   = (const float*)d_in[4];
    const float* Ds       = (const float*)d_in[5];
    const float* conv_w   = (const float*)d_in[6];
    const float* conv_b   = (const float*)d_in[7];
    const float* ln_gamma = (const float*)d_in[8];
    const float* ln_beta  = (const float*)d_in[9];
    const int*   order    = (const int*)d_in[10];
    const int*   inverse  = (const int*)d_in[11];
    const int*   pidx     = (const int*)d_in[12];
    const int*   vidx     = (const int*)d_in[13];
    const int    total    = in_sizes[13];   // 13096
    float* out = (float*)d_out;

    // workspace layout (all regions 256B-aligned)
    char* ws = (char*)d_ws;
    _Float16* wall_sw = (_Float16*)(ws);                  //  86 KB  (A-frag layout)
    _Float16* xs_sw   = (_Float16*)(ws + 131072u);        // 6.29 MB (B-frag layout)
    _Float16* xs_lin  = (_Float16*)(ws + 6422528u);       // 6.29 MB (linear, for conv)
    float*    dbl     = (float*)(ws + 12713984u);         // 14.7 MB (224 x BM)
    float*    u       = (float*)(ws + 27394048u);         // 12.6 MB
    float*    ys      = (float*)(ws + 39976960u);         // 12.6 MB

    // K0: combined projection matrix (224 x 192), A-fragment swizzled
    k_prep_wall<<<dim3((MROW * DCH + 255) / 256), dim3(256), 0, stream>>>(xpw, dtw, wall_sw);

    // K1: gather + f16 convert (linear + B-fragment swizzled)
    k_gather<<<dim3(BM / 256, DCH), dim3(256), 0, stream>>>(x, order, pidx, xs_lin, xs_sw, total);

    // K2: WMMA projection GEMM (224 x 192) x (192 x 16384), all-resident operands
    k_wmma_gemm<<<dim3(BM / (16 * NTPW * WVS), MROW / 16), dim3(WVS * 32), 0, stream>>>(
        wall_sw, xs_sw, dbl);

    // K3: depthwise conv + softplus(delta)
    k_conv_softplus<<<dim3(BM / 256, DCH), dim3(256), 0, stream>>>(
        xs_lin, conv_w, conv_b, dt_bias, dbl, u);

    // K4: selective scan (768 sequences, 16 lanes each)
    k_scan<<<dim3((BSZ * DCH * NST + 255) / 256), dim3(256), 0, stream>>>(
        dbl, u, A_logs, Ds, ys);

    // K5: inverse scatter + LayerNorm
    k_finalize<<<dim3((total * 32 + 255) / 256), dim3(256), 0, stream>>>(
        ys, inverse, vidx, ln_gamma, ln_beta, out, total);
}
